// DScCP_57750130262055
// MI455X (gfx1250) — matrix operand order repaired and checked by reading, not compiled, not run on hardware
//
#include <hip/hip_runtime.h>
#include <math.h>

// ---------------------------------------------------------------------------
// DScCP unrolled Chambolle-Pock, MI455X (gfx1250).
//
// WMMA everywhere (v_wmma_f32_16x16x32_f16); all B operands pre-packed in the
// exact 32x16 f16 register layout. Power-iteration step is ONE fused kernel:
//   fwd conv (M=64, K=27->32, 4 WMMAs) -> in-register half-wave shuffle
//   (D-fragment -> B-fragment, 8x shfl_xor(16) per k-step) -> tap-stacked
//   adjoint 1x1 GEMM (M=27, K=64, 2+2 WMMAs) -> T27 planes,
// then a fused 9-tap gather + norm-partials kernel. Early stop is a
// device-side done flag (fixed 50 iters => deterministic graph).
// Layer path fuses conv with the hardtanh dual update (u in f16 panels).
// Workspace ~94 MB, L2-resident (192 MB L2).
// ---------------------------------------------------------------------------

#define HWPIX   262144            // 512*512
#define IMW     512
#define IMH     512
#define NLAYER  20
#define FCH     64
#define SIG2    0.0009f           // sigma^2, sigma = 0.03
#define TOLPI   1e-4f

typedef __attribute__((ext_vector_type(16))) _Float16 v16h;
typedef __attribute__((ext_vector_type(8)))  _Float16 v8h;
typedef __attribute__((ext_vector_type(8)))  float    v8f;

struct PIState { int done; float val; float inv; float tau; };

__global__ void k_zero(int* z) { if (threadIdx.x == 0) *z = 0; }

__global__ void k_init_state(PIState* st) {
    if (threadIdx.x == 0) { st->done = 0; st->val = 1.0f; st->inv = 0.0f; st->tau = 0.0f; }
}

// ---- 16-bit A-matrix (16x32) placement as packed here (contiguous k) ----------------------
// element (M in 0..15, K in 0..31):
//   lane    = M + 16*(K>>4)
//   halfidx = K&15
// pack buffer: ap[(group*S + s)*512 + lane*16 + halfidx]

// Forward conv weights: A[m=f, K=cin*9+dy*3+dx], 4 groups of 16, S=1.
__global__ void k_pack_fwd(const float* __restrict__ Wt, _Float16* __restrict__ ap) {
    int id = blockIdx.x * blockDim.x + threadIdx.x;
    if (id >= NLAYER * 4 * 16 * 32) return;
    int K = id & 31; int t = id >> 5;
    int M = t & 15;  t >>= 4;
    int g = t & 3;   t >>= 2;
    int k = t;
    float v = 0.0f;
    if (K < 27) {
        int cin = K / 9, tap = K % 9;
        int f = g * 16 + M;
        v = Wt[((k * FCH + f) * 3 + cin) * 9 + tap];
    }
    int lane = M + 16 * (K >> 4);
    int half = K & 15;
    ap[(size_t)(k * 4 + g) * 512 + lane * 16 + half] = (_Float16)v;
}

// Adjoint tap-stacked weights: A[m = (ey*3+ex)*3 + c, K = f] = w[f,c,2-ey,2-ex].
// 2 groups of 16 (rows 27..31 zero), S=2 k-steps (K=64 exact).
__global__ void k_pack_adj(const float* __restrict__ Wt, _Float16* __restrict__ ap) {
    int id = blockIdx.x * blockDim.x + threadIdx.x;
    if (id >= NLAYER * 2 * 2 * 16 * 32) return;
    int K = id & 31; int t = id >> 5;
    int M = t & 15;  t >>= 4;
    int s = t & 1;   t >>= 1;
    int g = t & 1;   t >>= 1;
    int k = t;
    int m  = g * 16 + M;               // 0..31, rows >=27 are padding
    int f  = s * 32 + K;               // 0..63
    float v = 0.0f;
    if (m < 27) {
        int tap = m / 3, c = m % 3;
        int ey = tap / 3, ex = tap % 3;
        v = Wt[((k * FCH + f) * 3 + c) * 9 + (2 - ey) * 3 + (2 - ex)];
    }
    int lane = M + 16 * (K >> 4);
    int half = K & 15;
    ap[(size_t)(k * 4 + g * 2 + s) * 512 + lane * 16 + half] = (_Float16)v;
}

// ---- im2col to B-panel layout, with optional 1/norm scale -----------------
// dst[p*32 + e] = src[cin, i+dy-1, j+dx-1] * scale, e = cin*9+dy*3+dx (<27).
__global__ void k_im2col(const float* __restrict__ src, _Float16* __restrict__ dst,
                         const PIState* __restrict__ st, int useinv,
                         const int* __restrict__ guard)
{
    if (*guard) return;
    int p = blockIdx.x * 256 + threadIdx.x;          // grid covers HWPIX exactly
    int i = p >> 9, j = p & 511;
    float sc = useinv ? st->inv : 1.0f;
    _Float16 vals[32];
#pragma unroll
    for (int e = 0; e < 32; ++e) {
        float v = 0.0f;
        if (e < 27) {
            int cin = e / 9, tap = e % 9;            // constants after unroll
            int dy = tap / 3, dx = tap % 3;
            int r = i + dy - 1, c = j + dx - 1;
            if ((unsigned)r < (unsigned)IMH && (unsigned)c < (unsigned)IMW)
                v = src[(size_t)cin * HWPIX + r * IMW + c] * sc;
        }
        vals[e] = (_Float16)v;
    }
    v8h* out = (v8h*)(dst + (size_t)p * 32);
#pragma unroll
    for (int q = 0; q < 4; ++q) out[q] = *(const v8h*)(vals + q * 8);
}

// ---- fused power-iteration step: conv -> shuffle -> adjoint GEMM ----------
// 4 waves/block, 4 tiles/wave. Output: T27 f32 planes (tap-stacked adjoint).
__global__ void k_pi_step(const _Float16* __restrict__ pan, float* __restrict__ out,
                          const _Float16* __restrict__ apf,
                          const _Float16* __restrict__ apa,
                          const int* __restrict__ guard)
{
    if (*guard) return;
    int wv = threadIdx.x >> 5, l = threadIdx.x & 31;
    int tbase = (blockIdx.x * 4 + wv) * 4;
    int kq = l >> 4;

    v16h af[4], aa[4];
#pragma unroll
    for (int g = 0; g < 4; ++g) {
        af[g] = *(const v16h*)(apf + ((size_t)g * 32 + l) * 16);
        aa[g] = *(const v16h*)(apa + ((size_t)g * 32 + l) * 16);   // [g*2+s]
    }

    for (int t = 0; t < 4; ++t) {
        int tile = tbase + t;
        int row = tile >> 5, c0 = (tile & 31) << 4;
        int p = row * IMW + c0 + (l & 15);

        // forward conv: D[g] holds ch g*16+kq*8+r8 @ pixel l&15
        v16h b = *(const v16h*)(pan + (size_t)p * 32 + kq * 16);
        v8f zero = {};
        v8f acc[4];
#pragma unroll
        for (int g = 0; g < 4; ++g)
            acc[g] = __builtin_amdgcn_wmma_f32_16x16x32_f16(
                false, af[g], false, b, (short)0, zero, false, false);

        // adjoint 1x1 GEMM: B[h] = f16(ch s*32+kq*16+h @ pixel l&15),
        // assembled from D fragments via half-wave exchange.
        v8f acc0 = {}, acc1 = {};
#pragma unroll
        for (int s = 0; s < 2; ++s) {
            v16h bb;
#pragma unroll
            for (int r = 0; r < 8; ++r) {
                float own  = kq ? acc[2 * s + 1][r] : acc[2 * s][r];
                float send = kq ? acc[2 * s][r]     : acc[2 * s + 1][r];
                float oth  = __shfl_xor(send, 16, 32);
                bb[r]     = (_Float16)(kq ? oth : own);
                bb[8 + r] = (_Float16)(kq ? own : oth);
            }
            acc0 = __builtin_amdgcn_wmma_f32_16x16x32_f16(
                false, aa[s], false, bb, (short)0, acc0, false, false);
            acc1 = __builtin_amdgcn_wmma_f32_16x16x32_f16(
                false, aa[2 + s], false, bb, (short)0, acc1, false, false);
        }
#pragma unroll
        for (int r8 = 0; r8 < 8; ++r8) {
            int m0 = kq * 8 + r8;                    // group 0: rows 0..15
            out[(size_t)m0 * HWPIX + p] = acc0[r8];
            int m1 = 16 + kq * 8 + r8;               // group 1: keep <27
            if (m1 < 27) out[(size_t)m1 * HWPIX + p] = acc1[r8];
        }
    }
}

// ---- layer path: forward conv fused with dual hardtanh update -------------
// u[p][ch] = clip(u + (tau/mu)*conv_out, +-sigma^2), channel-major f16.
__global__ void k_conv_uupd(const _Float16* __restrict__ pan, _Float16* __restrict__ u,
                            const _Float16* __restrict__ ap,
                            const PIState* __restrict__ st,
                            const float* __restrict__ mu, int k,
                            const int* __restrict__ guard)
{
    if (*guard) return;
    float cf = st->tau / mu[k];
    int wv = threadIdx.x >> 5, l = threadIdx.x & 31;
    int tbase = (blockIdx.x * 4 + wv) * 4;
    int kq = l >> 4;

    v16h a[4];
#pragma unroll
    for (int g = 0; g < 4; ++g)
        a[g] = *(const v16h*)(ap + ((size_t)g * 32 + l) * 16);

    for (int t = 0; t < 4; ++t) {
        int tile = tbase + t;
        int row = tile >> 5, c0 = (tile & 31) << 4;
        int p = row * IMW + c0 + (l & 15);
        v16h b = *(const v16h*)(pan + (size_t)p * 32 + kq * 16);
        v8f zero = {};
        v8f acc[4];
#pragma unroll
        for (int g = 0; g < 4; ++g)
            acc[g] = __builtin_amdgcn_wmma_f32_16x16x32_f16(
                false, a[g], false, b, (short)0, zero, false, false);
#pragma unroll
        for (int g = 0; g < 4; ++g) {
            _Float16* up = u + (size_t)p * 64 + g * 16 + kq * 8;
            v8h uo = *(const v8h*)up;
            v8h un;
#pragma unroll
            for (int r8 = 0; r8 < 8; ++r8) {
                float v = (float)uo[r8] + cf * acc[g][r8];
                un[r8] = (_Float16)fminf(fmaxf(v, -SIG2), SIG2);
            }
            *(v8h*)up = un;
        }
    }
}

// ---- raw forward conv (u-init only): writes channel-major f16 panel -------
__global__ void k_conv_wmma(const _Float16* __restrict__ pan, _Float16* __restrict__ out,
                            const _Float16* __restrict__ ap, const int* __restrict__ guard)
{
    if (*guard) return;
    int wv = threadIdx.x >> 5, l = threadIdx.x & 31;
    int tbase = (blockIdx.x * 4 + wv) * 4;
    int kq = l >> 4;
    v16h a[4];
#pragma unroll
    for (int g = 0; g < 4; ++g)
        a[g] = *(const v16h*)(ap + ((size_t)g * 32 + l) * 16);
    for (int t = 0; t < 4; ++t) {
        int tile = tbase + t;
        int row = tile >> 5, c0 = (tile & 31) << 4;
        int p = row * IMW + c0 + (l & 15);
        v16h b = *(const v16h*)(pan + (size_t)p * 32 + kq * 16);
        v8f zero = {};
        v8f acc[4];
#pragma unroll
        for (int g = 0; g < 4; ++g)
            acc[g] = __builtin_amdgcn_wmma_f32_16x16x32_f16(
                false, a[g], false, b, (short)0, zero, false, false);
#pragma unroll
        for (int g = 0; g < 4; ++g) {
            v8h o;
#pragma unroll
            for (int r8 = 0; r8 < 8; ++r8) o[r8] = (_Float16)acc[g][r8];
            *(v8h*)(out + (size_t)p * 64 + g * 16 + kq * 8) = o;
        }
    }
}

// ---- adjoint 1x1 GEMM from a channel-major f16 panel (layer path) ---------
__global__ void k_gemm_wmma(const _Float16* __restrict__ pan, float* __restrict__ out,
                            const _Float16* __restrict__ ap, const int* __restrict__ guard)
{
    if (*guard) return;
    int wv = threadIdx.x >> 5, l = threadIdx.x & 31;
    int tbase = (blockIdx.x * 4 + wv) * 4;
    int kq = l >> 4;
    v16h a[4];                                   // [g*2+s]
#pragma unroll
    for (int q = 0; q < 4; ++q)
        a[q] = *(const v16h*)(ap + ((size_t)q * 32 + l) * 16);
    for (int t = 0; t < 4; ++t) {
        int tile = tbase + t;
        int row = tile >> 5, c0 = (tile & 31) << 4;
        int p = row * IMW + c0 + (l & 15);
        v8f acc0 = {}, acc1 = {};
#pragma unroll
        for (int s = 0; s < 2; ++s) {
            v16h b = *(const v16h*)(pan + (size_t)p * 64 + s * 32 + kq * 16);
            acc0 = __builtin_amdgcn_wmma_f32_16x16x32_f16(
                false, a[s], false, b, (short)0, acc0, false, false);
            acc1 = __builtin_amdgcn_wmma_f32_16x16x32_f16(
                false, a[2 + s], false, b, (short)0, acc1, false, false);
        }
#pragma unroll
        for (int r8 = 0; r8 < 8; ++r8) {
            int m0 = kq * 8 + r8;
            out[(size_t)m0 * HWPIX + p] = acc0[r8];
            int m1 = 16 + kq * 8 + r8;
            if (m1 < 27) out[(size_t)m1 * HWPIX + p] = acc1[r8];
        }
    }
}

// ---- fused 9-tap shift-add + sum-of-squares partials ----------------------
// z[c,p] = sum_{ey,ex} tmp[(ey*3+ex)*3+c][i+ey-1, j+ex-1]; part[blk]=sum z^2.
__global__ void k_gather_reduce(const float* __restrict__ tmp, float* __restrict__ z,
                                float* __restrict__ part, const int* __restrict__ guard)
{
    if (*guard) return;
    int p = blockIdx.x * 256 + threadIdx.x;          // grid = HWPIX/256 = 1024
    int i = p >> 9, j = p & 511;
    float a0 = 0.0f, a1 = 0.0f, a2 = 0.0f;
#pragma unroll
    for (int ey = 0; ey < 3; ++ey) {
        int r = i + ey - 1;
        if ((unsigned)r >= (unsigned)IMH) continue;
#pragma unroll
        for (int ex = 0; ex < 3; ++ex) {
            int c = j + ex - 1;
            if ((unsigned)c >= (unsigned)IMW) continue;
            const float* base = tmp + (size_t)(ey * 3 + ex) * 3 * HWPIX + r * IMW + c;
            a0 += base[0];
            a1 += base[HWPIX];
            a2 += base[2 * HWPIX];
        }
    }
    z[p]             = a0;
    z[HWPIX + p]     = a1;
    z[2 * HWPIX + p] = a2;

    __shared__ float sm[256];
    sm[threadIdx.x] = a0 * a0 + a1 * a1 + a2 * a2;
    __syncthreads();
    for (int o = 128; o > 0; o >>= 1) {
        if (threadIdx.x < o) sm[threadIdx.x] += sm[threadIdx.x + o];
        __syncthreads();
    }
    if (threadIdx.x == 0) part[blockIdx.x] = sm[0];
}

// ---- standalone sum-of-squares reduction (xinit normalization) ------------
__global__ void k_reduce_sumsq(const float* __restrict__ in, float* __restrict__ part,
                               int n, const int* __restrict__ guard)
{
    if (*guard) return;
    __shared__ float sm[256];
    float s = 0.0f;
    for (int i = blockIdx.x * 256 + threadIdx.x; i < n; i += 256 * gridDim.x) {
        float v = in[i]; s += v * v;
    }
    sm[threadIdx.x] = s;
    __syncthreads();
    for (int o = 128; o > 0; o >>= 1) {
        if (threadIdx.x < o) sm[threadIdx.x] += sm[threadIdx.x + o];
        __syncthreads();
    }
    if (threadIdx.x == 0) part[blockIdx.x] = sm[0];
}

__global__ void k_finalize(const float* __restrict__ part, PIState* st, int mode)
{
    __shared__ float sm[256];
    sm[threadIdx.x] = part[threadIdx.x] + part[threadIdx.x + 256]
                    + part[threadIdx.x + 512] + part[threadIdx.x + 768];
    __syncthreads();
    for (int o = 128; o > 0; o >>= 1) {
        if (threadIdx.x < o) sm[threadIdx.x] += sm[threadIdx.x + o];
        __syncthreads();
    }
    if (threadIdx.x == 0) {
        float sum = sm[0];
        if (mode == 0) {                         // normalize xinit
            st->inv = 1.0f / sqrtf(sum);
            st->val = 1.0f;
            st->done = 0;
        } else if (!st->done) {                  // power-iteration step
            float nv  = sqrtf(sum);
            float rel = fabsf(nv - st->val) / st->val;
            st->val = nv;
            st->inv = 1.0f / nv;
            st->tau = 0.99f / nv;
            if (rel < TOLPI) st->done = 1;
        }
    }
}

// ---- CP layer elementwise updates -----------------------------------------
__global__ void k_xcomb(const float* __restrict__ xc, const float* __restrict__ xp,
                        float* __restrict__ out, const float* __restrict__ mu, int k, int n)
{
    float m = mu[k];
    float alpha = 1.0f / sqrtf(1.0f + 2.0f * m);       // GAMMA = 1
    int i = blockIdx.x * 256 + threadIdx.x;
    if (i < n) out[i] = (1.0f + alpha) * xc[i] - alpha * xp[i];
}

__global__ void k_xupd(const float* __restrict__ x0, const float* __restrict__ xc,
                       const float* __restrict__ adj, float* __restrict__ xn,
                       const float* __restrict__ mu, int k, int n)
{
    float m = mu[k];
    float a = m / (m + 1.0f), b = 1.0f / (1.0f + m);
    int i = blockIdx.x * 256 + threadIdx.x;
    if (i < n) {
        float v = a * x0[i] + b * xc[i] - a * adj[i];
        xn[i] = fminf(fmaxf(v, 0.0f), 1.0f);
    }
}

// ---------------------------------------------------------------------------
extern "C" void kernel_launch(void* const* d_in, const int* in_sizes, int n_in,
                              void* d_out, int out_size, void* d_ws, size_t ws_size,
                              hipStream_t stream)
{
    (void)in_sizes; (void)n_in; (void)out_size; (void)ws_size;
    const float* x     = (const float*)d_in[0];   // [1,3,512,512]
    const float* Wt    = (const float*)d_in[1];   // [20,64,3,3,3]
    const float* mu    = (const float*)d_in[2];   // [20]
    const float* xinit = (const float*)d_in[3];   // [20,1,3,512,512]

    char* ws = (char*)d_ws;
    size_t off = 0;
    auto take = [&](size_t bytes) -> char* {
        char* p = ws + off;
        off += (bytes + 255) & ~(size_t)255;
        return p;
    };
    _Float16* UT    = (_Float16*)take((size_t)HWPIX * FCH * 2);  // dual u, [p][64] f16
    _Float16* XTCOL = (_Float16*)take((size_t)HWPIX * 32 * 2);   // im2col B-panels
    float*    T27   = (float*)take((size_t)27 * HWPIX * 4);      // adjoint tap temp
    float*    XT    = (float*)take((size_t)3 * HWPIX * 4);       // xcomb scratch
    float*    X2    = (float*)take((size_t)3 * HWPIX * 4);       // adjoint output
    float*    XA    = (float*)take((size_t)3 * HWPIX * 4);
    float*    XB    = (float*)take((size_t)3 * HWPIX * 4);
    float*    XC    = (float*)take((size_t)3 * HWPIX * 4);
    _Float16* APF   = (_Float16*)take((size_t)NLAYER * 4 * 512 * 2);
    _Float16* APA   = (_Float16*)take((size_t)NLAYER * 4 * 512 * 2);
    float*    PART  = (float*)take(1024 * 4);
    PIState*  ST    = (PIState*)take(sizeof(PIState));
    int*      ZERO  = (int*)take(4);

    const int n3 = 3 * HWPIX;
    const int g3 = (n3 + 255) / 256;
    const int gpix = HWPIX / 256;                // 1024
    const int gconv = 1024;                      // 16384 tiles / (4 waves * 4 tiles)

    k_zero<<<1, 32, 0, stream>>>(ZERO);
    k_pack_fwd<<<(NLAYER * 4 * 16 * 32 + 255) / 256, 256, 0, stream>>>(Wt, APF);
    k_pack_adj<<<(NLAYER * 4 * 16 * 32 + 255) / 256, 256, 0, stream>>>(Wt, APA);

    // u = conv(x, W[0])  (raw, unclipped)
    k_im2col<<<gpix, 256, 0, stream>>>(x, XTCOL, ST, 0, ZERO);
    k_conv_wmma<<<gconv, 128, 0, stream>>>(XTCOL, UT, APF, ZERO);

    const float* xprev = x;
    const float* xcurr = x;
    float* bufs[3] = {XA, XB, XC};
    int nb = 0;

    for (int k = 0; k < NLAYER; ++k) {
        const _Float16* apf = APF + (size_t)k * 4 * 512;
        const _Float16* apa = APA + (size_t)k * 4 * 512;
        const int* done = &ST->done;             // device addr of flag (offset 0)

        // ---- power iteration on W_k^T W_k, early stop via done flag ----
        k_init_state<<<1, 32, 0, stream>>>(ST);
        k_reduce_sumsq<<<1024, 256, 0, stream>>>(xinit + (size_t)k * n3, PART, n3, ZERO);
        k_finalize<<<1, 256, 0, stream>>>(PART, ST, 0);
        k_im2col<<<gpix, 256, 0, stream>>>(xinit + (size_t)k * n3, XTCOL, ST, 1, ZERO);
        for (int it = 0; it < 50; ++it) {
            k_pi_step<<<gconv, 128, 0, stream>>>(XTCOL, T27, apf, apa, done);
            k_gather_reduce<<<gpix, 256, 0, stream>>>(T27, X2, PART, done);
            k_finalize<<<1, 256, 0, stream>>>(PART, ST, 1);
            k_im2col<<<gpix, 256, 0, stream>>>(X2, XTCOL, ST, 1, done);
        }

        // ---- CP layer forward ----
        float* xnext = bufs[nb]; nb = (nb + 1) % 3;
        k_xcomb<<<g3, 256, 0, stream>>>(xcurr, xprev, XT, mu, k, n3);
        k_im2col<<<gpix, 256, 0, stream>>>(XT, XTCOL, ST, 0, ZERO);
        k_conv_uupd<<<gconv, 128, 0, stream>>>(XTCOL, UT, apf, ST, mu, k, ZERO);
        k_gemm_wmma<<<gconv, 128, 0, stream>>>(UT, T27, apa, ZERO);
        k_gather_reduce<<<gpix, 256, 0, stream>>>(T27, X2, PART, ZERO);
        k_xupd<<<g3, 256, 0, stream>>>(x, xcurr, X2, xnext, mu, k, n3);
        xprev = xcurr; xcurr = xnext;
    }

    hipMemcpyAsync(d_out, xcurr, (size_t)n3 * 4, hipMemcpyDeviceToDevice, stream);
}
